// ConvolutionalAttention_86947317940340
// MI455X (gfx1250) — matrix operand order repaired and checked by reading, not compiled, not run on hardware
//
#include <hip/hip_runtime.h>
#include <hip/hip_bf16.h>

#define DI __device__ __forceinline__

typedef __bf16 bf16x16 __attribute__((ext_vector_type(16)));
typedef __bf16 bf16x2  __attribute__((ext_vector_type(2)));
typedef float  f32x8   __attribute__((ext_vector_type(8)));
typedef int    v4i     __attribute__((ext_vector_type(4)));

struct alignas(16) U4 { unsigned int x, y, z, w; };
struct alignas(16) F4 { float x, y, z, w; };
struct alignas(8)  F2 { float x, y; };

union Frag {
    U4 q[2];
    unsigned int u[8];
    bf16x16 v;
};

#if __has_builtin(__builtin_amdgcn_global_load_async_to_lds_b128) && \
    __has_builtin(__builtin_amdgcn_s_wait_asynccnt)
#define USE_ASYNC_LDS 1
#else
#define USE_ASYNC_LDS 0
#endif

// packed f32x2 -> bf16x2 (round-to-nearest-even)
static DI unsigned int pk(float a, float b) {
#if __has_builtin(__builtin_amdgcn_cvt_pk_bf16_f32)
    union { bf16x2 h; unsigned int u; } cv;
    cv.h = __builtin_amdgcn_cvt_pk_bf16_f32(a, b);
    return cv.u;
#else
    union { __hip_bfloat162 h; unsigned int u; } cv;
    cv.h = __float22bfloat162_rn(make_float2(a, b));
    return cv.u;
#endif
}
static DI unsigned short f2bf(float f) {
    unsigned int x = __float_as_uint(f);
    x += 0x7fffu + ((x >> 16) & 1u);
    return (unsigned short)(x >> 16);
}
static DI f32x8 zero8() {
    f32x8 z = {0.f, 0.f, 0.f, 0.f, 0.f, 0.f, 0.f, 0.f};
    return z;
}
static DI f32x8 wmma_bf16(const Frag& a, const Frag& b, f32x8 c) {
    // D = A(16x32 bf16) * B(32x16 bf16) + C(16x16 f32)
    return __builtin_amdgcn_wmma_f32_16x16x32_bf16(false, a.v, false, b.v,
                                                   (short)0, c, false, false);
}
// A fragment from a row-major bf16-pair image (pitch in dwords, mult of 4):
// u[0..3] = row[base .. base+3], u[4..7] = row[base+8 .. base+11]
static DI void load_a(Frag& f, const unsigned int* img, int row, int pitch,
                      int kk, int hi) {
    const unsigned int* p = img + row * pitch + kk * 16 + hi * 4;
    f.q[0] = *(const U4*)(p);
    f.q[1] = *(const U4*)(p + 8);
}
// B fragment: u[0..3] = row[base .. base+3], u[4..7] = row[base+8 .. base+11]
static DI void load_b(Frag& f, const unsigned int* img, int row, int pitch,
                      int kk, int hi) {
    const unsigned int* p = img + row * pitch + kk * 16 + hi * 4;
    f.q[0] = *(const U4*)(p);
    f.q[1] = *(const U4*)(p + 8);
}

static constexpr int CIN = 256, CK = 128, H = 128, W = 128, HW = H * W;
static constexpr int XP = 132;  // stage-1 x tile: [n][k-pair], dword pitch
static constexpr int QP = 68;   // stage-2 Q/K: [row][w-pair], dword pitch
static constexpr int VP = 68;   // stage-2 V: [w][j-pair], dword pitch
static constexpr int WPAIR = CIN / 2;           // 128 dwords per weight row
static constexpr int WMATD = CK * WPAIR;        // 16384 dwords per weight matrix

// ---------------------------------------------------------------------------
// Stage 0: one-shot weight conversion f32 -> bf16 pair image (L2-resident).
// ---------------------------------------------------------------------------
__global__ __launch_bounds__(256) void wconv_kernel(
    const float* __restrict__ wq, const float* __restrict__ wk,
    const float* __restrict__ wv, unsigned int* __restrict__ wbf)
{
    int idx = blockIdx.x * 256 + threadIdx.x;    // 3 * 16384 pairs
    int mat = idx >> 14;
    int p   = idx & (WMATD - 1);
    const float* wm = (mat == 0) ? wq : ((mat == 1) ? wk : wv);
    const F2 t = *(const F2*)(wm + 2 * p);
    wbf[idx] = pk(t.x, t.y);
}

// ---------------------------------------------------------------------------
// Stage 1: fused q/k/v 1x1-conv projections.
// One WG = (batch, 128-pixel tile); three GEMMs [128x256]x[256x128] in bf16
// WMMA; outputs stored bf16 in [b][c][p] order for stage 2.
// ---------------------------------------------------------------------------
__global__ __launch_bounds__(256) void qkv_kernel(
    const float* __restrict__ x, const unsigned int* __restrict__ wbf,
    const float* __restrict__ bq, const float* __restrict__ bk,
    const float* __restrict__ bv,
    unsigned short* __restrict__ qs, unsigned short* __restrict__ ks,
    unsigned short* __restrict__ vs)
{
    __shared__ unsigned int xlds[128 * XP];  // [n][k/2]: pair (k, k+1) of column n

    const int b    = blockIdx.x >> 7;
    const int p0   = (blockIdx.x & 127) * 128;
    const int lane = threadIdx.x & 31;
    const int wid  = threadIdx.x >> 5;
    const int hi   = lane >> 4;
    const int l15  = lane & 15;

    // Prefetch the L2-resident bf16 weight image (global_prefetch_b8).
    __builtin_prefetch(wbf + threadIdx.x * 192, 0, 3);

    // Cooperative load: x tile [256 x 128] f32 -> bf16 pairs, transposed to
    // [n][k-pair] so B fragments are contiguous dwords.
    const float* xb = x + (size_t)b * CIN * HW + p0;
#pragma unroll
    for (int it = 0; it < 16; ++it) {
        int idx = it * 256 + threadIdx.x;   // 128 k-pairs * 32 float4 n-blocks
        int pi  = idx >> 5;
        int nb  = idx & 31;
        const F4 lo = ((const F4*)(xb + (size_t)(2 * pi)     * HW))[nb];
        const F4 hv = ((const F4*)(xb + (size_t)(2 * pi + 1) * HW))[nb];
        int n0 = nb * 4;
        xlds[(n0 + 0) * XP + pi] = pk(lo.x, hv.x);
        xlds[(n0 + 1) * XP + pi] = pk(lo.y, hv.y);
        xlds[(n0 + 2) * XP + pi] = pk(lo.z, hv.z);
        xlds[(n0 + 3) * XP + pi] = pk(lo.w, hv.w);
    }
    __syncthreads();

    const float* bvecs[3]   = {bq, bk, bv};
    unsigned short* outs[3] = {qs, ks, vs};

    const int m0 = wid * 16;    // this wave's 16-row output-channel strip

#pragma unroll
    for (int mat = 0; mat < 3; ++mat) {
        const unsigned int* wmat = wbf + mat * WMATD;
        f32x8 acc[8];
#pragma unroll
        for (int nt = 0; nt < 8; ++nt) acc[nt] = zero8();

        for (int kk = 0; kk < 8; ++kk) {      // K = 256 in steps of 32
            Frag a;                           // weight strip: two global b128
            load_a(a, wmat, m0 + l15, WPAIR, kk, hi);
#pragma unroll
            for (int nt = 0; nt < 8; ++nt) {
                Frag bf;
                load_b(bf, xlds, nt * 16 + l15, XP, kk, hi);
                acc[nt] = wmma_bf16(a, bf, acc[nt]);
            }
        }

        // Epilogue: + bias, f32 -> bf16, store [b][c][p].
        const float* bb = bvecs[mat];
        unsigned short* op = outs[mat] + (size_t)b * CK * HW + p0;
#pragma unroll
        for (int r = 0; r < 8; ++r) {
            int   row  = m0 + r + hi * 8;
            float bias = bb[row];
#pragma unroll
            for (int nt = 0; nt < 8; ++nt) {
                float val = acc[nt][r] + bias;
                op[(size_t)row * HW + nt * 16 + l15] = f2bf(val);
            }
        }
    }
}

// ---------------------------------------------------------------------------
// Stage 2: per (b,c) attention over rows.  One WG per (b,c).
//   S = Q K^T / sqrt(128); P = softmax_j(S); O = P V
// ---------------------------------------------------------------------------
__global__ __launch_bounds__(256) void attn_kernel(
    const unsigned short* __restrict__ qs, const unsigned short* __restrict__ ks,
    const unsigned short* __restrict__ vs, float* __restrict__ out)
{
    __shared__ unsigned int Qs[128 * QP];   // [i][w-pair], later [i][j-pair] (P)
    __shared__ unsigned int Ks[128 * QP];   // [j][w-pair]
    __shared__ unsigned int Vs[128 * VP];   // [w][j-pair] (transposed for B op)

    const int bc   = blockIdx.x;            // b*128 + c
    const int lane = threadIdx.x & 31;
    const int wid  = threadIdx.x >> 5;
    const int hi   = lane >> 4;
    const int l15  = lane & 15;

    const U4* qsrc = (const U4*)(qs + (size_t)bc * HW);
    const U4* ksrc = (const U4*)(ks + (size_t)bc * HW);
    const unsigned short* vsrc = vs + (size_t)bc * HW;

    // Q and K: contiguous 32 KB each -> pitched LDS rows.
#if USE_ASYNC_LDS
    // gfx1250 async DMA path: global_load_async_to_lds_b128, ASYNCcnt tracked.
#pragma unroll
    for (int it = 0; it < 8; ++it) {
        int idx = it * 256 + threadIdx.x;            // 2048 16B chunks
        int r  = idx >> 4;                           // 16 chunks per 128-elem row
        int wp = (idx & 15) * 4;
        __builtin_amdgcn_global_load_async_to_lds_b128(
            (v4i*)(qsrc + idx), (v4i*)&Qs[r * QP + wp], 0, 0);
        __builtin_amdgcn_global_load_async_to_lds_b128(
            (v4i*)(ksrc + idx), (v4i*)&Ks[r * QP + wp], 0, 0);
    }
#else
#pragma unroll
    for (int it = 0; it < 8; ++it) {
        int idx = it * 256 + threadIdx.x;            // 2048 16B chunks
        U4 dq = qsrc[idx];
        U4 dk = ksrc[idx];
        int r  = idx >> 4;
        int wp = (idx & 15) * 4;
        *(U4*)&Qs[r * QP + wp] = dq;
        *(U4*)&Ks[r * QP + wp] = dk;
    }
#endif
    // V: gather j-pairs, store transposed [w][j-pair].
#pragma unroll
    for (int it = 0; it < 32; ++it) {
        int idx = it * 256 + threadIdx.x;            // 64 j-pairs * 128 w
        int pi = idx >> 7;
        int w  = idx & 127;
        unsigned int lo = vsrc[(2 * pi)     * W + w];
        unsigned int hv = vsrc[(2 * pi + 1) * W + w];
        Vs[w * VP + pi] = lo | (hv << 16);
    }
#if USE_ASYNC_LDS
    __builtin_amdgcn_s_wait_asynccnt(0);   // own async LDS writes done
#endif
    __syncthreads();

    const int i0 = wid * 16;                // this wave's 16-row query strip

    // ---- S = Q K^T -------------------------------------------------------
    f32x8 acc[8];
#pragma unroll
    for (int jt = 0; jt < 8; ++jt) acc[jt] = zero8();

    for (int kk = 0; kk < 4; ++kk) {        // contraction over W=128
        Frag a;
        load_a(a, Qs, i0 + l15, QP, kk, hi);
#pragma unroll
        for (int jt = 0; jt < 8; ++jt) {
            Frag bf;                        // row j of K supplies column j of S
            load_b(bf, Ks, jt * 16 + l15, QP, kk, hi);
            acc[jt] = wmma_bf16(a, bf, acc[jt]);
        }
    }

    // ---- softmax over j --------------------------------------------------
    const float kExpScale = 1.4426950408889634f * 0.08838834764831845f; // log2e/sqrt(128)
#pragma unroll
    for (int r = 0; r < 8; ++r) {
        float m = -3.0e38f;
#pragma unroll
        for (int jt = 0; jt < 8; ++jt) m = fmaxf(m, acc[jt][r]);
        m = fmaxf(m, __shfl_xor(m, 1, 32));
        m = fmaxf(m, __shfl_xor(m, 2, 32));
        m = fmaxf(m, __shfl_xor(m, 4, 32));
        m = fmaxf(m, __shfl_xor(m, 8, 32));
        float s = 0.f;
#pragma unroll
        for (int jt = 0; jt < 8; ++jt) {
            float e = exp2f((acc[jt][r] - m) * kExpScale);
            acc[jt][r] = e;
            s += e;
        }
        s += __shfl_xor(s, 1, 32);
        s += __shfl_xor(s, 2, 32);
        s += __shfl_xor(s, 4, 32);
        s += __shfl_xor(s, 8, 32);
        float inv = 1.0f / s;
#pragma unroll
        for (int jt = 0; jt < 8; ++jt) acc[jt][r] *= inv;
    }

    // ---- write P over this wave's own Q rows (per-wave DS ordering) ------
    unsigned short* Ps = (unsigned short*)Qs;
#pragma unroll
    for (int r = 0; r < 8; ++r) {
        int i = i0 + r + hi * 8;
#pragma unroll
        for (int jt = 0; jt < 8; ++jt)
            Ps[i * (QP * 2) + jt * 16 + l15] = f2bf(acc[jt][r]);
    }

    // ---- O = P V ---------------------------------------------------------
    f32x8 o[8];
#pragma unroll
    for (int wt = 0; wt < 8; ++wt) o[wt] = zero8();

    for (int kk = 0; kk < 4; ++kk) {        // contraction over j=128
        Frag a;
        load_a(a, Qs, i0 + l15, QP, kk, hi);
#pragma unroll
        for (int wt = 0; wt < 8; ++wt) {
            Frag bf;
            load_b(bf, Vs, wt * 16 + l15, VP, kk, hi);
            o[wt] = wmma_bf16(a, bf, o[wt]);
        }
    }

    float* op = out + (size_t)bc * HW;
#pragma unroll
    for (int r = 0; r < 8; ++r) {
        int i = i0 + r + hi * 8;
#pragma unroll
        for (int wt = 0; wt < 8; ++wt)
            op[i * W + wt * 16 + l15] = o[wt][r];
    }
}

// ---------------------------------------------------------------------------
extern "C" void kernel_launch(void* const* d_in, const int* in_sizes, int n_in,
                              void* d_out, int out_size, void* d_ws, size_t ws_size,
                              hipStream_t stream) {
    const float* x  = (const float*)d_in[0];
    const float* wq = (const float*)d_in[1];
    const float* bq = (const float*)d_in[2];
    const float* wk = (const float*)d_in[3];
    const float* bk = (const float*)d_in[4];
    const float* wv = (const float*)d_in[5];
    const float* bv = (const float*)d_in[6];
    float* out = (float*)d_out;

    const size_t per = (size_t)16 * CK * HW;          // 33,554,432 bf16 elems
    unsigned short* qsv = (unsigned short*)d_ws;
    unsigned short* ksv = qsv + per;
    unsigned short* vsv = ksv + per;
    unsigned int*   wbf = (unsigned int*)(vsv + per); // 3*16384 dwords (192 KB)

    // Stage 0: weights f32 -> bf16 pair image (3*16384 pairs, 192 blocks)
    wconv_kernel<<<192, 256, 0, stream>>>(wq, wk, wv, wbf);
    // Stage 1: 16 batches * 128 pixel-tiles
    qkv_kernel<<<2048, 256, 0, stream>>>(x, wbf, bq, bk, bv, qsv, ksv, vsv);
    // Stage 2: 16 batches * 128 channels
    attn_kernel<<<2048, 256, 0, stream>>>(qsv, ksv, vsv, out);
}
